// DilatedResidualBlock_4234837753927
// MI455X (gfx1250) — matrix operand order repaired and checked by reading, not compiled, not run on hardware
//
#include <hip/hip_runtime.h>
#include <hip/hip_bf16.h>

// ---------------------------------------------------------------------------
// DilatedResidualBlock for MI455X (gfx1250): wave32 + WMMA f16 (f32 accum),
// async global->LDS gathers (ASYNCcnt) and TDM tensor_load_to_lds staging.
// Structure exploited: dst[e] == e/16  =>  per-node fused attention with
// M=16 WMMA tiles; biases before BatchNorm cancel and are skipped.
// ---------------------------------------------------------------------------

typedef __attribute__((ext_vector_type(16))) _Float16 v16h;
typedef __attribute__((ext_vector_type(8)))  _Float16 v8h;
typedef __attribute__((ext_vector_type(8)))  float    v8f;
typedef __attribute__((ext_vector_type(4)))  unsigned u32x4;
typedef __attribute__((ext_vector_type(8)))  int      i32x8;
typedef __attribute__((ext_vector_type(4)))  int      i32x4;

static constexpr int kN   = 50000;
static constexpr int kK   = 16;
static constexpr int kE   = kN * kK;      // 800000
static constexpr int kDIN = 128;
static constexpr int kDO  = 256;

#if defined(__has_builtin)
#if __has_builtin(__builtin_amdgcn_tensor_load_to_lds)
#define HAVE_TDM 1
#endif
#endif

// generic->LDS byte offset: low 32 bits of a generic shared pointer are the
// group-segment offset on AMDGPU (aperture in the high bits).
__device__ inline unsigned lds_addr_of(const void* p) {
  return (unsigned)(unsigned long long)p;
}

// GLOBAL_LOAD_ASYNC_TO_LDS_B128: per-lane dsaddr (VDST) <- per-lane global (VADDR)
__device__ inline void async_gather_b128(unsigned lds_off, const void* gptr) {
  asm volatile("global_load_async_to_lds_b128 %0, %1, off"
               :: "v"(lds_off), "v"((unsigned long long)gptr) : "memory");
}

__device__ inline void wait_async0() {
#if defined(__has_builtin) && __has_builtin(__builtin_amdgcn_s_wait_asynccnt)
  __builtin_amdgcn_s_wait_asynccnt(0);
#else
  asm volatile("s_wait_asynccnt 0" ::: "memory");
#endif
}

__device__ inline void wait_tensor0() {
#if defined(__has_builtin) && __has_builtin(__builtin_amdgcn_s_wait_tensorcnt)
  __builtin_amdgcn_s_wait_tensorcnt(0);
#else
  asm volatile("s_wait_tensorcnt 0" ::: "memory");
#endif
}

// TDM: DMA `ndw` dwords from global `ga` to LDS offset `ldsa` (1-row 2D tile,
// data_size=4B, D# type=2).  ndw must fit tile_dim0 (16 bits).
__device__ inline void tdm_copy_dwords(unsigned ldsa, unsigned long long ga, int ndw) {
#if defined(HAVE_TDM)
  u32x4 g0 = {1u,                                   // count=1 (valid descriptor)
              ldsa,                                 // lds_addr
              (unsigned)ga,                         // global_addr[31:0]
              (unsigned)((ga >> 32) & 0x1FFFFFFu) | (2u << 30)}; // addr[56:32] | type=2
  i32x8 g1 = {(int)(2u << 16),                      // workgroup_mask=0, data_size=2 (4B)
              (int)((unsigned)ndw << 16),           // tensor_dim0[15:0] @ bits 63:48
              (int)(((unsigned)ndw >> 16) | (1u << 16)), // tensor_dim0[31:16] | tensor_dim1=1
              (int)((unsigned)ndw << 16),           // tile_dim0 @ bits 127:112
              1,                                    // tile_dim1=1, tile_dim2=0
              ndw,                                  // tensor_dim0_stride[31:0]
              0, 0};
  i32x4 z4 = {0, 0, 0, 0};
#if __clang_major__ >= 23
  i32x8 z8 = {0, 0, 0, 0, 0, 0, 0, 0};
  __builtin_amdgcn_tensor_load_to_lds(g0, g1, z4, z4, z8, 0);
#else
  __builtin_amdgcn_tensor_load_to_lds(g0, g1, z4, z4, 0);
#endif
#else
  (void)ldsa; (void)ga; (void)ndw;
#endif
}

// ---------------- weight packing into WMMA B-matrix lane layout -------------
__global__ void pack_b(const float* __restrict__ W, int Kd, int Co, _Float16* __restrict__ out) {
  int total = Kd * Co;
  int Kt = Kd / 32;
  for (int p = blockIdx.x * blockDim.x + threadIdx.x; p < total; p += gridDim.x * blockDim.x) {
    int idx  = p & 15;
    int lane = (p >> 4) & 31;
    int kt   = (p >> 9) % Kt;
    int nt   = p / (512 * Kt);
    int row  = kt * 32 + (lane >> 4) * 8 + (idx & 7) + (idx >> 3) * 16;
    int col  = nt * 16 + (lane & 15);
    out[p] = (_Float16)W[row * Co + col];
  }
}

__global__ void f32_to_f16(const float* __restrict__ in, _Float16* __restrict__ out, int n) {
  for (int i = blockIdx.x * blockDim.x + threadIdx.x; i < n; i += gridDim.x * blockDim.x)
    out[i] = (_Float16)in[i];
}

// ---------------- A-fragment loader (16-bit A 16x32 ISA layout) -------------
__device__ inline v16h load_a_g(const _Float16* __restrict__ A, int lda, int rbase, int kbase, int lane) {
  int m = lane & 15, half = lane >> 4;
  const _Float16* p = A + (size_t)(rbase + m) * lda + kbase + half * 8;
  v16h a;
  *((v8h*)&a)       = *((const v8h*)p);
  *(((v8h*)&a) + 1) = *((const v8h*)(p + 16));
  return a;
}

// ------- WMMA GEMM, NTW n-tiles per wave (A-fragment reuse) -----------------
template <int BK, int CO, int NTW>
__global__ void gemm_wmma(const _Float16* __restrict__ A, const _Float16* __restrict__ Bp,
                          float* __restrict__ out, int Mtiles) {
  int lane = threadIdx.x & 31;
  int mt = blockIdx.x * 8 + (threadIdx.x >> 5);
  if (mt >= Mtiles) return;                       // whole-wave uniform exit
  int nt0 = blockIdx.y * NTW;
  constexpr int Kt = BK / 32;
  int rbase = mt * 16;
  v8f acc[NTW];
#pragma unroll
  for (int j = 0; j < NTW; ++j) acc[j] = v8f{0.f, 0.f, 0.f, 0.f, 0.f, 0.f, 0.f, 0.f};
#pragma unroll
  for (int kt = 0; kt < Kt; ++kt) {
    __builtin_prefetch(A + (size_t)rbase * BK + (kt + 1) * 32, 0, 3);
    v16h a = load_a_g(A, BK, rbase, kt * 32, lane);
#pragma unroll
    for (int j = 0; j < NTW; ++j) {
      v16h b = *(const v16h*)(Bp + ((size_t)((nt0 + j) * Kt + kt) * 32 + lane) * 16);
      acc[j] = __builtin_amdgcn_wmma_f32_16x16x32_f16(false, a, false, b, (short)0, acc[j],
                                                      false, false);
    }
  }
  int n = lane & 15, half = lane >> 4;
#pragma unroll
  for (int j = 0; j < NTW; ++j) {
    float* o = out + (size_t)rbase * CO + (nt0 + j) * 16 + n;
#pragma unroll
    for (int v = 0; v < 8; ++v) o[(size_t)(v + 8 * half) * CO] = acc[j][v];
  }
}

// ---------------- relative position encoder (pre-BN, f16 out) ---------------
template <int CH>
__global__ void enc_pre(const float* __restrict__ pos, const int* __restrict__ src,
                        const int* __restrict__ dst, const float* __restrict__ encW,
                        _Float16* __restrict__ out) {
  __shared__ float w[10 * CH];
  for (int i = threadIdx.x; i < 10 * CH; i += blockDim.x) w[i] = encW[i];
  __syncthreads();
  int e = blockIdx.x * blockDim.x + threadIdx.x;
  if (e >= kE) return;
  int j = src[e], i = dst[e];
  float pix = pos[3 * i], piy = pos[3 * i + 1], piz = pos[3 * i + 2];
  float pjx = pos[3 * j], pjy = pos[3 * j + 1], pjz = pos[3 * j + 2];
  float dx = pjx - pix, dy = pjy - piy, dz = pjz - piz;
  float dist = sqrtf(dx * dx + dy * dy + dz * dz);
  float rel[10] = {pix, piy, piz, pjx, pjy, pjz, dx, dy, dz, dist};
  _Float16* o = out + (size_t)e * CH;
#pragma unroll 4
  for (int c = 0; c < CH; ++c) {
    float a = 0.f;
#pragma unroll
    for (int k = 0; k < 10; ++k) a += rel[k] * w[k * CH + c];
    o[c] = (_Float16)a;
  }
}

// ---------------- two-stage deterministic BatchNorm stats -------------------
template <typename T>
__global__ void bn_stats(const T* __restrict__ x, int rows, int C, float* __restrict__ partial) {
  int c = threadIdx.x;
  float s = 0.f, ss = 0.f;
  for (int r = blockIdx.x; r < rows; r += gridDim.x) {
    float v = (float)x[(size_t)r * C + c];
    s += v; ss += v * v;
  }
  partial[(size_t)(blockIdx.x * 2 + 0) * C + c] = s;
  partial[(size_t)(blockIdx.x * 2 + 1) * C + c] = ss;
}

__global__ void bn_finalize(const float* __restrict__ partial, int chunks, int C,
                            const float* __restrict__ g, const float* __restrict__ be,
                            float rinv, float* __restrict__ ss) {
  int c = threadIdx.x;
  float s = 0.f, q = 0.f;
  for (int k = 0; k < chunks; ++k) {
    s += partial[(size_t)(k * 2 + 0) * C + c];
    q += partial[(size_t)(k * 2 + 1) * C + c];
  }
  float mu = s * rinv, var = q * rinv - mu * mu;
  float sc = g[c] * rsqrtf(var + 1e-6f);
  ss[c] = sc;
  ss[C + c] = be[c] - mu * sc;   // bias cancels in BN: not added upstream
}

template <bool LRELU, typename TIN, typename TOUT>
__global__ void bn_apply(const TIN* __restrict__ x, const float* __restrict__ ss, int C,
                         size_t total, TOUT* __restrict__ out) {
  for (size_t i = blockIdx.x * (size_t)blockDim.x + threadIdx.x; i < total;
       i += (size_t)gridDim.x * blockDim.x) {
    int c = (int)(i % C);
    float y = (float)x[i] * ss[c] + ss[C + c];
    if (LRELU) y = (y >= 0.f) ? y : 0.2f * y;
    out[i] = (TOUT)y;
  }
}

// ---------------- fused per-node local feature attention (WMMA) -------------
// feat[16,C] = [ hprev[src_m] | enc[e] ] gathered via async global->LDS;
// att_W staged in LDS once per block via TDM; att = feat @ attW; per-column
// softmax over the 16 edge rows; agg[node] = sum_m scores*feat.
template <int C>
__global__ void lfa_attn(const _Float16* __restrict__ hprev,   // [N, C/2]
                         const _Float16* __restrict__ enc,     // [E, C/2]
                         const int* __restrict__ srcIdx,       // [E]
                         const _Float16* __restrict__ attWp,   // packed [C,C]
                         _Float16* __restrict__ agg) {         // [N, C]
  constexpr int HC = C / 2, Kt = C / 32, Nt = C / 16, CHR = C / 8;
  __shared__ _Float16 sfeat[8 * 16 * C];
  __shared__ _Float16 sB[C * C];
  int lane = threadIdx.x & 31;
  int wv = threadIdx.x >> 5;
  int node = blockIdx.x * 8 + wv;                  // N % 8 == 0, no guard
  _Float16* feat = sfeat + wv * 16 * C;
  int e0 = node * 16;
  // async gather of the feat tile into LDS (16B chunks, per-lane addresses)
#pragma unroll
  for (int ch = lane; ch < 2 * C; ch += 32) {
    int m = ch / CHR, q = ch % CHR;
    const _Float16* sp;
    if (q < CHR / 2) { int s = srcIdx[e0 + m]; sp = hprev + (size_t)s * HC + q * 8; }
    else             { sp = enc + (size_t)(e0 + m) * HC + (q - CHR / 2) * 8; }
    async_gather_b128(lds_addr_of(feat + m * C + q * 8), sp);
  }
  // stage att_W panel in LDS: TDM DMA if available, else cooperative copy
#if defined(HAVE_TDM)
  if (threadIdx.x == 0)
    tdm_copy_dwords(lds_addr_of(sB), (unsigned long long)attWp, C * C * 2 / 4);
  if (threadIdx.x < 32) wait_tensor0();
#else
  for (int i = threadIdx.x; i < C * C / 8; i += blockDim.x)
    ((v8h*)sB)[i] = ((const v8h*)attWp)[i];
#endif
  wait_async0();
  __syncthreads();

  int r = lane & 15, half = lane >> 4;
  v16h afrag[Kt];
#pragma unroll
  for (int kt = 0; kt < Kt; ++kt) {
    const _Float16* p = feat + r * C + kt * 32 + half * 8;
    *((v8h*)&afrag[kt])       = *((const v8h*)p);
    *(((v8h*)&afrag[kt]) + 1) = *((const v8h*)(p + 16));
  }

#pragma unroll
  for (int nt = 0; nt < Nt; ++nt) {
    v8f acc = {0.f, 0.f, 0.f, 0.f, 0.f, 0.f, 0.f, 0.f};
#pragma unroll
    for (int kt = 0; kt < Kt; ++kt) {
      v16h b = *(const v16h*)(sB + ((size_t)(nt * Kt + kt) * 32 + lane) * 16);
      acc = __builtin_amdgcn_wmma_f32_16x16x32_f16(false, afrag[kt], false, b, (short)0, acc,
                                                   false, false);
    }
    float mx = acc[0];
#pragma unroll
    for (int v = 1; v < 8; ++v) mx = fmaxf(mx, acc[v]);
    mx = fmaxf(mx, __shfl_xor(mx, 16, 32));
    float ex[8], s = 0.f;
#pragma unroll
    for (int v = 0; v < 8; ++v) { ex[v] = __expf(acc[v] - mx); s += ex[v]; }
    s += __shfl_xor(s, 16, 32);
    float inv = 1.f / (s + 1e-16f);
    float part = 0.f;
#pragma unroll
    for (int v = 0; v < 8; ++v)
      part += ex[v] * inv * (float)feat[(v + 8 * half) * C + nt * 16 + r];
    part += __shfl_xor(part, 16, 32);
    if (half == 0) agg[(size_t)node * C + nt * 16 + r] = (_Float16)part;
  }
}

// ---------------- output fusion + tail ---------------------------------------
__global__ void final_fuse(const float* __restrict__ m2p, const float* __restrict__ ss2,
                           const float* __restrict__ scp, const float* __restrict__ ss1,
                           float* __restrict__ out) {
  size_t total = (size_t)kN * kDO;
  for (size_t i = blockIdx.x * (size_t)blockDim.x + threadIdx.x; i < total;
       i += (size_t)gridDim.x * blockDim.x) {
    int c = (int)(i % kDO);
    float y = m2p[i] * ss2[c] + ss2[kDO + c] + scp[i] * ss1[c] + ss1[kDO + c];
    out[i] = (y >= 0.f) ? y : 0.2f * y;
  }
}

__global__ void copy_tail(const float* __restrict__ pos, const int* __restrict__ batch,
                          float* __restrict__ out) {
  int i = blockIdx.x * blockDim.x + threadIdx.x;
  if (i < kN * 3) out[i] = pos[i];
  else if (i < kN * 4) out[i] = __int_as_float(batch[i - kN * 3]);
}

// ---------------------------------------------------------------------------
extern "C" void kernel_launch(void* const* d_in, const int* in_sizes, int n_in,
                              void* d_out, int out_size, void* d_ws, size_t ws_size,
                              hipStream_t stream) {
  (void)in_sizes; (void)n_in; (void)out_size; (void)ws_size;
  // Input order: x, pos, batch, edge_index, then params via sorted-key tree flatten.
  const float* x    = (const float*)d_in[0];
  const float* pos  = (const float*)d_in[1];
  const int*   bat  = (const int*)d_in[2];
  const int*   src  = (const int*)d_in[3];           // edge_index[0]
  const int*   dst  = (const int*)d_in[3] + kE;      // edge_index[1]
  const float* a1_attW  = (const float*)d_in[4];
  const float* a1_encW  = (const float*)d_in[5];
  const float* a1_encBe = (const float*)d_in[7];
  const float* a1_encG  = (const float*)d_in[8];
  const float* a1_postW = (const float*)d_in[9];
  const float* a1_postBe= (const float*)d_in[11];
  const float* a1_postG = (const float*)d_in[12];
  const float* a2_attW  = (const float*)d_in[13];
  const float* a2_encW  = (const float*)d_in[14];
  const float* a2_encBe = (const float*)d_in[16];
  const float* a2_encG  = (const float*)d_in[17];
  const float* a2_postW = (const float*)d_in[18];
  const float* a2_postBe= (const float*)d_in[20];
  const float* a2_postG = (const float*)d_in[21];
  const float* m1W  = (const float*)d_in[22];
  const float* m1Be = (const float*)d_in[24];
  const float* m1G  = (const float*)d_in[25];
  const float* m2W  = (const float*)d_in[26];
  const float* m2Be = (const float*)d_in[28];
  const float* m2G  = (const float*)d_in[29];
  const float* scW  = (const float*)d_in[30];
  const float* scBe = (const float*)d_in[32];
  const float* scG  = (const float*)d_in[33];

  // ---- workspace bump allocation ----
  char* ws = (char*)d_ws;
  size_t off = 0;
  auto alloc = [&](size_t bytes) { char* p = ws + off; off = (off + bytes + 255) & ~(size_t)255; return p; };
  _Float16* x16   = (_Float16*)alloc((size_t)kN * kDIN * 2);
  _Float16* pSc   = (_Float16*)alloc(128 * 256 * 2);
  _Float16* pM1   = (_Float16*)alloc(128 * 32 * 2);
  _Float16* pM2   = (_Float16*)alloc(128 * 256 * 2);
  _Float16* pAtt1 = (_Float16*)alloc(64 * 64 * 2);
  _Float16* pPost1= (_Float16*)alloc(64 * 64 * 2);
  _Float16* pAtt2 = (_Float16*)alloc(128 * 128 * 2);
  _Float16* pPost2= (_Float16*)alloc(128 * 128 * 2);
  float*    scp   = (float*)alloc((size_t)kN * 256 * 4);
  float*    m1p   = (float*)alloc((size_t)kN * 32 * 4);
  _Float16* h1    = (_Float16*)alloc((size_t)kN * 32 * 2);
  _Float16* encp  = (_Float16*)alloc((size_t)kE * 64 * 2);   // pre-BN (reused lfa1/lfa2)
  _Float16* encA  = (_Float16*)alloc((size_t)kE * 64 * 2);   // post-BN (reused)
  _Float16* agg   = (_Float16*)alloc((size_t)kN * 128 * 2);  // reused lfa1/lfa2
  float*    postp = (float*)alloc((size_t)kN * 128 * 4);     // reused
  _Float16* h2    = (_Float16*)alloc((size_t)kN * 64 * 2);
  _Float16* h3    = (_Float16*)alloc((size_t)kN * 128 * 2);
  float*    m2p   = (float*)alloc((size_t)kN * 256 * 4);
  float*    part  = (float*)alloc((size_t)512 * 2 * 256 * 4);
  float*    ssSc  = (float*)alloc(2 * 256 * 4);
  float*    ssM1  = (float*)alloc(2 * 256 * 4);
  float*    ssE1  = (float*)alloc(2 * 256 * 4);
  float*    ssP1  = (float*)alloc(2 * 256 * 4);
  float*    ssE2  = (float*)alloc(2 * 256 * 4);
  float*    ssP2  = (float*)alloc(2 * 256 * 4);
  float*    ssM2  = (float*)alloc(2 * 256 * 4);

  const int Mt = kN / 16;                 // 3125
  const int gBx = (Mt + 7) / 8;           // 391
  const float rinvN = 1.0f / kN, rinvE = 1.0f / kE;

  // ---- pack weights / convert activations ----
  hipLaunchKernelGGL(pack_b, dim3(128), dim3(256), 0, stream, scW, 128, 256, pSc);
  hipLaunchKernelGGL(pack_b, dim3(16),  dim3(256), 0, stream, m1W, 128, 32,  pM1);
  hipLaunchKernelGGL(pack_b, dim3(128), dim3(256), 0, stream, m2W, 128, 256, pM2);
  hipLaunchKernelGGL(pack_b, dim3(16),  dim3(256), 0, stream, a1_attW, 64, 64, pAtt1);
  hipLaunchKernelGGL(pack_b, dim3(16),  dim3(256), 0, stream, a1_postW, 64, 64, pPost1);
  hipLaunchKernelGGL(pack_b, dim3(64),  dim3(256), 0, stream, a2_attW, 128, 128, pAtt2);
  hipLaunchKernelGGL(pack_b, dim3(64),  dim3(256), 0, stream, a2_postW, 128, 128, pPost2);
  hipLaunchKernelGGL(f32_to_f16, dim3(4096), dim3(256), 0, stream, x, x16, kN * kDIN);

  // ---- shortcut GEMM + stats ----
  hipLaunchKernelGGL((gemm_wmma<128, 256, 4>), dim3(gBx, 4), dim3(256), 0, stream, x16, pSc, scp, Mt);
  hipLaunchKernelGGL(bn_stats<float>, dim3(256), dim3(256), 0, stream, scp, kN, 256, part);
  hipLaunchKernelGGL(bn_finalize, dim3(1), dim3(256), 0, stream, part, 256, 256, scG, scBe, rinvN, ssSc);

  // ---- m1 -> h1 ----
  hipLaunchKernelGGL((gemm_wmma<128, 32, 2>), dim3(gBx, 1), dim3(256), 0, stream, x16, pM1, m1p, Mt);
  hipLaunchKernelGGL(bn_stats<float>, dim3(256), dim3(32), 0, stream, m1p, kN, 32, part);
  hipLaunchKernelGGL(bn_finalize, dim3(1), dim3(32), 0, stream, part, 256, 32, m1G, m1Be, rinvN, ssM1);
  hipLaunchKernelGGL((bn_apply<true, float, _Float16>), dim3(4096), dim3(256), 0, stream,
                     m1p, ssM1, 32, (size_t)kN * 32, h1);

  // ---- LFA 1 (C=64) ----
  hipLaunchKernelGGL(enc_pre<32>, dim3((kE + 255) / 256), dim3(256), 0, stream, pos, src, dst, a1_encW, encp);
  hipLaunchKernelGGL(bn_stats<_Float16>, dim3(512), dim3(32), 0, stream, encp, kE, 32, part);
  hipLaunchKernelGGL(bn_finalize, dim3(1), dim3(32), 0, stream, part, 512, 32, a1_encG, a1_encBe, rinvE, ssE1);
  hipLaunchKernelGGL((bn_apply<true, _Float16, _Float16>), dim3(4096), dim3(256), 0, stream,
                     encp, ssE1, 32, (size_t)kE * 32, encA);
  hipLaunchKernelGGL(lfa_attn<64>, dim3(kN / 8), dim3(256), 0, stream, h1, encA, src, pAtt1, agg);
  hipLaunchKernelGGL((gemm_wmma<64, 64, 4>), dim3(gBx, 1), dim3(256), 0, stream, agg, pPost1, postp, Mt);
  hipLaunchKernelGGL(bn_stats<float>, dim3(256), dim3(64), 0, stream, postp, kN, 64, part);
  hipLaunchKernelGGL(bn_finalize, dim3(1), dim3(64), 0, stream, part, 256, 64, a1_postG, a1_postBe, rinvN, ssP1);
  hipLaunchKernelGGL((bn_apply<true, float, _Float16>), dim3(4096), dim3(256), 0, stream,
                     postp, ssP1, 64, (size_t)kN * 64, h2);

  // ---- LFA 2 (C=128) ----
  hipLaunchKernelGGL(enc_pre<64>, dim3((kE + 255) / 256), dim3(256), 0, stream, pos, src, dst, a2_encW, encp);
  hipLaunchKernelGGL(bn_stats<_Float16>, dim3(512), dim3(64), 0, stream, encp, kE, 64, part);
  hipLaunchKernelGGL(bn_finalize, dim3(1), dim3(64), 0, stream, part, 512, 64, a2_encG, a2_encBe, rinvE, ssE2);
  hipLaunchKernelGGL((bn_apply<true, _Float16, _Float16>), dim3(4096), dim3(256), 0, stream,
                     encp, ssE2, 64, (size_t)kE * 64, encA);
  hipLaunchKernelGGL(lfa_attn<128>, dim3(kN / 8), dim3(256), 0, stream, h2, encA, src, pAtt2, agg);
  hipLaunchKernelGGL((gemm_wmma<128, 128, 4>), dim3(gBx, 2), dim3(256), 0, stream, agg, pPost2, postp, Mt);
  hipLaunchKernelGGL(bn_stats<float>, dim3(256), dim3(128), 0, stream, postp, kN, 128, part);
  hipLaunchKernelGGL(bn_finalize, dim3(1), dim3(128), 0, stream, part, 256, 128, a2_postG, a2_postBe, rinvN, ssP2);
  hipLaunchKernelGGL((bn_apply<true, float, _Float16>), dim3(4096), dim3(256), 0, stream,
                     postp, ssP2, 128, (size_t)kN * 128, h3);

  // ---- m2 + residual fuse ----
  hipLaunchKernelGGL((gemm_wmma<128, 256, 4>), dim3(gBx, 4), dim3(256), 0, stream, h3, pM2, m2p, Mt);
  hipLaunchKernelGGL(bn_stats<float>, dim3(256), dim3(256), 0, stream, m2p, kN, 256, part);
  hipLaunchKernelGGL(bn_finalize, dim3(1), dim3(256), 0, stream, part, 256, 256, m2G, m2Be, rinvN, ssM2);
  hipLaunchKernelGGL(final_fuse, dim3(4096), dim3(256), 0, stream, m2p, ssM2, scp, ssSc, (float*)d_out);
  hipLaunchKernelGGL(copy_tail, dim3((kN * 4 + 255) / 256), dim3(256), 0, stream,
                     pos, bat, (float*)d_out + (size_t)kN * kDO);
}
